// EncoderGRU_16509854286338
// MI455X (gfx1250) — matrix operand order repaired and checked by reading, not compiled, not run on hardware
//
#include <hip/hip_runtime.h>

// ---------------------------------------------------------------------------
// 2-layer GRU (B=128, T=512, D=128, H=512), batch_first, eval mode.
//  - bf16 WMMA (v_wmma_f32_16x16x32_bf16) for all matmuls, fp32 accumulate.
//  - W tiles staged into LDS via async global->LDS DMA (ASYNCcnt path),
//    WMMA B operands read back with ds_load_b128 (8 waves share one N-tile).
//  - Software-pipelined layers: one launch per slot s runs layer0@t=s
//    (blocks 0..31) and layer1@t=s-1 (blocks 32..63). 513 launches total.
//  - h double-buffered per layer (fp32 master + bf16 shadow for WMMA A).
// ---------------------------------------------------------------------------

typedef __attribute__((ext_vector_type(16))) __bf16 v16bf;
typedef __attribute__((ext_vector_type(8)))  float  v8f;

#define B_   128
#define T_   512
#define D_   128
#define H_   512
#define G3H_ 1536

__device__ __forceinline__ unsigned short f2bf(float f) {
    unsigned int u = __float_as_uint(f);
    unsigned int r = u + 0x7FFFu + ((u >> 16) & 1u);   // round-to-nearest-even
    return (unsigned short)(r >> 16);
}

// ---- generic f32 -> bf16 conversion (weights, x) --------------------------
__global__ void k_f32_to_bf16(const float* __restrict__ in,
                              unsigned short* __restrict__ out, int n) {
    int i = blockIdx.x * blockDim.x + threadIdx.x;
    int stride = gridDim.x * blockDim.x;
    for (; i < n; i += stride) out[i] = f2bf(in[i]);
}

// ---- h init: fp32 copy + bf16 shadow --------------------------------------
__global__ void k_init_h(const float* __restrict__ in,
                         float* __restrict__ outf,
                         unsigned short* __restrict__ outb, int n) {
    int i = blockIdx.x * blockDim.x + threadIdx.x;
    int stride = gridDim.x * blockDim.x;
    for (; i < n; i += stride) {
        float v = in[i];
        outf[i] = v;
        outb[i] = f2bf(v);
    }
}

// ---- final hT copy --------------------------------------------------------
__global__ void k_copy_f32(const float* __restrict__ in,
                           float* __restrict__ out, int n) {
    int i = blockIdx.x * blockDim.x + threadIdx.x;
    int stride = gridDim.x * blockDim.x;
    for (; i < n; i += stride) out[i] = in[i];
}

// ---- async global -> LDS, 16 bytes per lane (gfx1250 ASYNCcnt path) -------
__device__ __forceinline__ void async_b128_to_lds(unsigned int lds_byte_off,
                                                  const void* gptr) {
    asm volatile("global_load_async_to_lds_b128 %0, %1, off"
                 :: "v"(lds_byte_off),
                    "v"((unsigned long long)(size_t)gptr)
                 : "memory");
}

// ---------------------------------------------------------------------------
// One GRU timestep for a 16-column slice (n0..n0+15) of one layer.
//   gi = x_t @ W_ih^T + b_ih ; gh = h @ W_hh^T + b_hh (kept separate: n-gate
//   needs r * gh_n).  r = sig(gi_r+gh_r); z = sig(gi_z+gh_z);
//   n = tanh(gi_n + r*gh_n); h' = (1-z)*n + z*h.
// Block = 256 threads = 8 waves; wave w owns M-tile rows [16w,16w+16).
// W tiles (3 gates x 16 cols x K) are staged in LDS, shared by all 8 waves.
// ---------------------------------------------------------------------------
template <int KIN>
__device__ __forceinline__ void gru_tile_step(
    const unsigned short* __restrict__ A, int lda,      // bf16 [128,KIN]
    const unsigned short* __restrict__ Wih,             // bf16 [3H,KIN]
    const unsigned short* __restrict__ Whh,             // bf16 [3H,H]
    const float* __restrict__ b_ih, const float* __restrict__ b_hh,
    const float* __restrict__ hprev_f,                  // f32  [128,512]
    const unsigned short* __restrict__ hprev_b,         // bf16 [128,512]
    float* __restrict__ hnew_f, unsigned short* __restrict__ hnew_b,
    unsigned short* __restrict__ out_b,                 // optional bf16 [128,512]
    float* __restrict__ out_f, int out_f_rs,            // optional f32
    int n0) {
    extern __shared__ unsigned short smem[];            // AS3 kept for ds_load
    unsigned short* wihL = smem;                        // [3*16, KIN]
    unsigned short* whhL = smem + 3 * 16 * KIN;         // [3*16, 512]

    const int tid  = threadIdx.x;
    const int lane = tid & 31;
    const int m0   = (tid >> 5) * 16;                   // wave -> M tile
    const int kh   = lane >> 4;                         // K-half selector
    const int l15  = lane & 15;

    // ---- stage W tiles into LDS with async DMA (bypasses VGPRs) -----------
    {
        const unsigned int base_ih = (unsigned int)(size_t)wihL;
        const unsigned int base_hh = (unsigned int)(size_t)whhL;
        constexpr int QI = 3 * 16 * KIN / 8;            // 16B quads, /256 = exact
        for (int q = tid; q < QI; q += 256) {
            const int row = q / (KIN / 8), kq = q % (KIN / 8);
            const int g = row >> 4, r = row & 15;
            async_b128_to_lds(base_ih + (unsigned int)(row * KIN + kq * 8) * 2u,
                              Wih + (size_t)(g * H_ + n0 + r) * KIN + kq * 8);
        }
        constexpr int QH = 3 * 16 * H_ / 8;
        for (int q = tid; q < QH; q += 256) {
            const int row = q / (H_ / 8), kq = q % (H_ / 8);
            const int g = row >> 4, r = row & 15;
            async_b128_to_lds(base_hh + (unsigned int)(row * H_ + kq * 8) * 2u,
                              Whh + (size_t)(g * H_ + n0 + r) * H_ + kq * 8);
        }
        asm volatile("s_wait_asynccnt 0" ::: "memory");
        __syncthreads();
    }

    v8f accI[3] = {{}, {}, {}};
    v8f accH[3] = {{}, {}, {}};

    // ---- input-gate GEMM: A tiles (global) x Wih tiles (LDS) --------------
    {
        const unsigned short* arow = A + (size_t)(m0 + l15) * lda;
#pragma unroll 4
        for (int k0 = 0; k0 < KIN; k0 += 32) {
            union { v16bf v; uint4 q[2]; } ua;
            // A 16x32 bf16: lanes 0-15 K{0..7,16..23}, lanes 16-31 K{8..15,24..31}
            ua.q[0] = *(const uint4*)(arow + k0 + kh * 8);
            ua.q[1] = *(const uint4*)(arow + k0 + 16 + kh * 8);
#pragma unroll
            for (int g = 0; g < 3; ++g) {
                // B 32x16 bf16: lane = column, lanes 0-15 K{0..7,16..23}, lanes 16-31 K{8..15,24..31}
                const unsigned short* brow =
                    wihL + (g * 16 + l15) * KIN + k0 + kh * 8;
                union { v16bf v; uint4 q[2]; } ub;
                ub.q[0] = *(const uint4*)(brow);
                ub.q[1] = *(const uint4*)(brow + 16);
                accI[g] = __builtin_amdgcn_wmma_f32_16x16x32_bf16(
                    false, ua.v, false, ub.v, (short)0, accI[g], false, false);
            }
        }
    }

    // ---- hidden-gate GEMM: h tiles (global) x Whh tiles (LDS), K = 512 ----
    {
        const unsigned short* arow = hprev_b + (size_t)(m0 + l15) * H_;
#pragma unroll 4
        for (int k0 = 0; k0 < H_; k0 += 32) {
            union { v16bf v; uint4 q[2]; } ua;
            ua.q[0] = *(const uint4*)(arow + k0 + kh * 8);
            ua.q[1] = *(const uint4*)(arow + k0 + 16 + kh * 8);
#pragma unroll
            for (int g = 0; g < 3; ++g) {
                const unsigned short* brow =
                    whhL + (g * 16 + l15) * H_ + k0 + kh * 8;
                union { v16bf v; uint4 q[2]; } ub;
                ub.q[0] = *(const uint4*)(brow);
                ub.q[1] = *(const uint4*)(brow + 16);
                accH[g] = __builtin_amdgcn_wmma_f32_16x16x32_bf16(
                    false, ua.v, false, ub.v, (short)0, accH[g], false, false);
            }
        }
    }

    // ---- gate math + blend.  C/D layout: elem r -> (M = kh*8+r, N = l15) --
    const int n = n0 + l15;
    const float bir = b_ih[n],           bhr = b_hh[n];
    const float biz = b_ih[H_ + n],      bhz = b_hh[H_ + n];
    const float bin_ = b_ih[2 * H_ + n], bhn = b_hh[2 * H_ + n];

#pragma unroll
    for (int r = 0; r < 8; ++r) {
        const int m = m0 + kh * 8 + r;
        const float gi_r = accI[0][r] + bir;
        const float gi_z = accI[1][r] + biz;
        const float gi_n = accI[2][r] + bin_;
        const float gh_r = accH[0][r] + bhr;
        const float gh_z = accH[1][r] + bhz;
        const float gh_n = accH[2][r] + bhn;

        const float rg = 1.0f / (1.0f + __expf(-(gi_r + gh_r)));
        const float zg = 1.0f / (1.0f + __expf(-(gi_z + gh_z)));
        const float ng = tanhf(gi_n + rg * gh_n);

        const float hp = hprev_f[(size_t)m * H_ + n];
        const float hn = (1.0f - zg) * ng + zg * hp;

        hnew_f[(size_t)m * H_ + n] = hn;
        const unsigned short hb = f2bf(hn);
        hnew_b[(size_t)m * H_ + n] = hb;
        if (out_b) out_b[(size_t)m * H_ + n] = hb;
        if (out_f) out_f[(size_t)m * out_f_rs + n] = hn;
    }
}

// ---------------------------------------------------------------------------
// One pipeline slot: blocks 0..31 run layer0 @ t=s, blocks 32..63 run
// layer1 @ t=s-1 (independent: layer1 consumes o0 written in slot s-1).
// ---------------------------------------------------------------------------
__global__ __launch_bounds__(256)
void k_gru_slot(int do_l0, int do_l1,
                const unsigned short* __restrict__ A0,   // xb + s*D
                const unsigned short* __restrict__ Wih0,
                const unsigned short* __restrict__ Whh0,
                const float* __restrict__ bih0, const float* __restrict__ bhh0,
                const float* __restrict__ h0pf, const unsigned short* __restrict__ h0pb,
                float* __restrict__ h0nf, unsigned short* __restrict__ h0nb,
                unsigned short* __restrict__ o0w,        // layer0 out (bf16)
                const unsigned short* __restrict__ o0r,  // layer1 in  (bf16)
                const unsigned short* __restrict__ Wih1,
                const unsigned short* __restrict__ Whh1,
                const float* __restrict__ bih1, const float* __restrict__ bhh1,
                const float* __restrict__ h1pf, const unsigned short* __restrict__ h1pb,
                float* __restrict__ h1nf, unsigned short* __restrict__ h1nb,
                float* __restrict__ out1t) {             // out + t1*H
    const int layer = blockIdx.x >> 5;
    const int n0 = (blockIdx.x & 31) * 16;
    if (layer == 0) {
        if (!do_l0) return;
        gru_tile_step<D_>(A0, T_ * D_, Wih0, Whh0, bih0, bhh0,
                          h0pf, h0pb, h0nf, h0nb,
                          o0w, nullptr, 0, n0);
    } else {
        if (!do_l1) return;
        gru_tile_step<H_>(o0r, H_, Wih1, Whh1, bih1, bhh1,
                          h1pf, h1pb, h1nf, h1nb,
                          nullptr, out1t, T_ * H_, n0);
    }
}

// ---------------------------------------------------------------------------
extern "C" void kernel_launch(void* const* d_in, const int* in_sizes, int n_in,
                              void* d_out, int out_size, void* d_ws, size_t ws_size,
                              hipStream_t stream) {
    (void)in_sizes; (void)n_in; (void)out_size; (void)ws_size;

    const float* x    = (const float*)d_in[0]; // [B,T,D]
    const float* h    = (const float*)d_in[1]; // [2,B,H]
    const float* Wih0 = (const float*)d_in[2]; // [3H,D]
    const float* Whh0 = (const float*)d_in[3]; // [3H,H]
    const float* bih0 = (const float*)d_in[4];
    const float* bhh0 = (const float*)d_in[5];
    const float* Wih1 = (const float*)d_in[6]; // [3H,H]
    const float* Whh1 = (const float*)d_in[7]; // [3H,H]
    const float* bih1 = (const float*)d_in[8];
    const float* bhh1 = (const float*)d_in[9];
    float* out = (float*)d_out; // out1 [B,T,H] ++ hT [2,B,H]

    // ---- workspace carve-up (256B aligned) --------------------------------
    char* w = (char*)d_ws;
    auto carve = [&](size_t bytes) -> void* {
        void* p = (void*)w;
        w += (bytes + 255) & ~(size_t)255;
        return p;
    };
    unsigned short* xb    = (unsigned short*)carve((size_t)B_ * T_ * D_ * 2);
    unsigned short* Wih0b = (unsigned short*)carve((size_t)G3H_ * D_ * 2);
    unsigned short* Whh0b = (unsigned short*)carve((size_t)G3H_ * H_ * 2);
    unsigned short* Wih1b = (unsigned short*)carve((size_t)G3H_ * H_ * 2);
    unsigned short* Whh1b = (unsigned short*)carve((size_t)G3H_ * H_ * 2);
    float*          h0f[2]; unsigned short* h0b[2];
    float*          h1f[2]; unsigned short* h1b[2];
    for (int i = 0; i < 2; ++i) {
        h0f[i] = (float*)carve((size_t)B_ * H_ * 4);
        h0b[i] = (unsigned short*)carve((size_t)B_ * H_ * 2);
        h1f[i] = (float*)carve((size_t)B_ * H_ * 4);
        h1b[i] = (unsigned short*)carve((size_t)B_ * H_ * 2);
    }
    unsigned short* o0b[2];
    o0b[0] = (unsigned short*)carve((size_t)B_ * H_ * 2);
    o0b[1] = (unsigned short*)carve((size_t)B_ * H_ * 2);

    // ---- one-time conversions --------------------------------------------
    auto cvt = [&](const float* src, unsigned short* dst, int n) {
        int blocks = (n + 255) / 256;
        if (blocks > 2048) blocks = 2048;
        k_f32_to_bf16<<<blocks, 256, 0, stream>>>(src, dst, n);
    };
    cvt(x,    xb,    B_ * T_ * D_);
    cvt(Wih0, Wih0b, G3H_ * D_);
    cvt(Whh0, Whh0b, G3H_ * H_);
    cvt(Wih1, Wih1b, G3H_ * H_);
    cvt(Whh1, Whh1b, G3H_ * H_);

    k_init_h<<<(B_ * H_ + 255) / 256, 256, 0, stream>>>(h,           h0f[0], h0b[0], B_ * H_);
k_init_h<<<(B_ * H_ + 255) / 256, 256, 0, stream>>>(h + B_ * H_, h1f[0], h1b[0], B_ * H_);

    // ---- pipelined recurrence: slot s = layer0@t=s + layer1@t=s-1 ---------
    const size_t LDS_BYTES = (size_t)(3 * 16 * H_ + 3 * 16 * H_) * 2; // 96 KB
    for (int s = 0; s <= T_; ++s) {
        const int t1 = (s >= 1) ? (s - 1) : 0;
        k_gru_slot<<<64, 256, LDS_BYTES, stream>>>(
            (s < T_) ? 1 : 0, (s >= 1) ? 1 : 0,
            xb + (size_t)((s < T_) ? s : 0) * D_,
            Wih0b, Whh0b, bih0, bhh0,
            h0f[s & 1], h0b[s & 1], h0f[(s + 1) & 1], h0b[(s + 1) & 1],
            o0b[s & 1],
            o0b[t1 & 1],
            Wih1b, Whh1b, bih1, bhh1,
            h1f[t1 & 1], h1b[t1 & 1], h1f[(t1 + 1) & 1], h1b[(t1 + 1) & 1],
            out + (size_t)t1 * H_);
    }

    // ---- final hidden states: after t=511 both layers' h live in buffer 0 -
    float* hT = out + (size_t)B_ * T_ * H_;
    k_copy_f32<<<(B_ * H_ + 255) / 256, 256, 0, stream>>>(h0f[0], hT,            B_ * H_);
    k_copy_f32<<<(B_ * H_ + 255) / 256, 256, 0, stream>>>(h1f[0], hT + B_ * H_, B_ * H_);
}
